// MultiheadSelfAttention_3865470566617
// MI455X (gfx1250) — matrix operand order repaired and checked by reading, not compiled, not run on hardware
//
#include <hip/hip_runtime.h>
#include <math.h>

typedef __attribute__((ext_vector_type(16))) __bf16 v16bf;
typedef __attribute__((ext_vector_type(8)))  __bf16 v8bf;
typedef __attribute__((ext_vector_type(4)))  __bf16 v4bf;
typedef __attribute__((ext_vector_type(8)))  float  v8f;
typedef int v4i_vs __attribute__((vector_size(16)));   // matches builtin pointee

#define DHID 2048   // model dim = row stride everywhere

#if __has_builtin(__builtin_amdgcn_global_load_async_to_lds_b128)
#define HAS_ASYNC 1
#else
#define HAS_ASYNC 0
#endif

static __device__ __forceinline__ v8f wmma_bf16(v16bf a, v16bf b, v8f c) {
  return __builtin_amdgcn_wmma_f32_16x16x32_bf16(false, a, false, b, (short)0, c, false, false);
}
static __device__ __forceinline__ v16bf cat16(v8bf lo, v8bf hi) {
  return __builtin_shufflevector(lo, hi, 0,1,2,3,4,5,6,7,8,9,10,11,12,13,14,15);
}
static __device__ __forceinline__ v8bf cvt8(v8f x) {
  v8bf r;
  #pragma unroll
  for (int i = 0; i < 8; ++i) r[i] = (__bf16)x[i];
  return r;
}

// 16-byte global -> LDS copy: async path if available, else via VGPRs.
static __device__ __forceinline__ void copy16(const __bf16* g, __bf16* l) {
#if HAS_ASYNC
  __builtin_amdgcn_global_load_async_to_lds_b128(
      (__attribute__((address_space(1))) v4i_vs*)(void*)g,
      (__attribute__((address_space(3))) v4i_vs*)(void*)l, 0, 0);
#else
  *(v8bf*)l = *(const v8bf*)g;
#endif
}
// Wait until at most N async-to-LDS instructions remain in flight (in-order).
static __device__ __forceinline__ void async_wait_all() {
#if HAS_ASYNC
#if __has_builtin(__builtin_amdgcn_s_wait_asynccnt)
  __builtin_amdgcn_s_wait_asynccnt(0);
#else
  asm volatile("s_wait_asynccnt 0x0" ::: "memory");
#endif
#endif
}
static __device__ __forceinline__ void async_wait_4() {
#if HAS_ASYNC
#if __has_builtin(__builtin_amdgcn_s_wait_asynccnt)
  __builtin_amdgcn_s_wait_asynccnt(4);
#else
  asm volatile("s_wait_asynccnt 0x4" ::: "memory");
#endif
#endif
}

// ---------------------------------------------------------------------------
// Pre-pass: f32 -> bf16 elementwise (for x)
// ---------------------------------------------------------------------------
__global__ void cvt_f32_bf16(const float* __restrict__ in, __bf16* __restrict__ out, int n4) {
  int idx = blockIdx.x * blockDim.x + threadIdx.x;
  if (idx >= n4) return;
  float4 v = *(const float4*)(in + (size_t)idx * 4);
  v4bf o;
  o[0] = (__bf16)v.x; o[1] = (__bf16)v.y; o[2] = (__bf16)v.z; o[3] = (__bf16)v.w;
  *(v4bf*)(out + (size_t)idx * 4) = o;
}

// ---------------------------------------------------------------------------
// Pre-pass: W[N,K] f32 -> W^T[K,N] bf16 (32x32 LDS tile transpose)
// ---------------------------------------------------------------------------
__global__ __launch_bounds__(256) void transpose_bf16(const float* __restrict__ w,
                                                      __bf16* __restrict__ wt) {
  __shared__ float t[32][33];
  int k0 = blockIdx.x * 32, n0 = blockIdx.y * 32;
  int tx = threadIdx.x & 31, ty = threadIdx.x >> 5;   // 32 x 8
  #pragma unroll
  for (int i = 0; i < 4; ++i)
    t[ty + i * 8][tx] = w[(size_t)(n0 + ty + i * 8) * DHID + k0 + tx];
  __syncthreads();
  #pragma unroll
  for (int i = 0; i < 4; ++i)
    wt[(size_t)(k0 + ty + i * 8) * DHID + n0 + tx] = (__bf16)t[tx][ty + i * 8];
}

// ---------------------------------------------------------------------------
// GEMM: C[M,2048] = A[M,2048](bf16) @ BT[2048,2048](bf16, pre-transposed W).
// Block 128x128, K-step 32; 8 waves 2x4, each 64x32.
// Double-buffered async LDS staging (4 async instrs per wave per tile).
// ---------------------------------------------------------------------------
static __device__ __forceinline__ void stage_ab(const __bf16* __restrict__ A,
                                                const __bf16* __restrict__ BT,
                                                int m0, int n0, int k0,
                                                __bf16* dA, __bf16* dB, int tid) {
  #pragma unroll
  for (int it = 0; it < 2; ++it) {               // A tile: 512 x 16B chunks
    int idx = tid + it * 256;
    int r = idx >> 2, c = idx & 3;
    copy16(A + (size_t)(m0 + r) * DHID + k0 + c * 8, &dA[r * 40 + c * 8]);
  }
  #pragma unroll
  for (int it = 0; it < 2; ++it) {               // B tile: 512 x 16B chunks
    int idx = tid + it * 256;
    int r = idx >> 4, c = idx & 15;
    copy16(BT + (size_t)(k0 + r) * DHID + n0 + c * 8, &dB[r * 136 + c * 8]);
  }
}

template <bool OUT_BF16>
__global__ __launch_bounds__(256) void gemm_TN(const __bf16* __restrict__ A,
                                               const __bf16* __restrict__ BT,
                                               void* __restrict__ Cout) {
  __shared__ __align__(16) __bf16 sA[2][128 * 40];   // 2 x 10240 B
  __shared__ __align__(16) __bf16 sB[2][32 * 136];   // 2 x  8704 B
  const int tid = threadIdx.x, lane = tid & 31, wave = tid >> 5;
  const int wm = wave >> 2, wn = wave & 3;
  const int half = lane >> 4, l16 = lane & 15;
  const int m0 = blockIdx.y * 128, n0 = blockIdx.x * 128;

  v8f acc[4][2] = {};

  stage_ab(A, BT, m0, n0, 0, sA[0], sB[0], tid);     // prologue: tile 0

  for (int k0 = 0; k0 < DHID; k0 += 32) {
    const int cur = (k0 >> 5) & 1;
    if (k0 + 32 < DHID) {                            // prefetch next tile
      stage_ab(A, BT, m0, n0, k0 + 32, sA[cur ^ 1], sB[cur ^ 1], tid);
      async_wait_4();                                // tile k0 done (in-order)
    } else {
      async_wait_all();
    }
    __syncthreads();

    v16bf af[4];
    #pragma unroll
    for (int mt = 0; mt < 4; ++mt) {
      const __bf16* base = &sA[cur][(wm * 64 + mt * 16 + l16) * 40 + half * 8];
      af[mt] = cat16(*(const v8bf*)base, *(const v8bf*)(base + 16));
    }
    #pragma unroll
    for (int nt = 0; nt < 2; ++nt) {
      const __bf16* bb = &sB[cur][(half * 8) * 136 + wn * 32 + nt * 16 + l16];
      v16bf bf;
      #pragma unroll
      for (int s = 0; s < 16; ++s) bf[s] = bb[((s & 7) + (s >> 3) * 16) * 136];
      #pragma unroll
      for (int mt = 0; mt < 4; ++mt)
        acc[mt][nt] = wmma_bf16(af[mt], bf, acc[mt][nt]);
    }
    __syncthreads();                                 // buf reusable next iter
  }

  #pragma unroll
  for (int mt = 0; mt < 4; ++mt)
    #pragma unroll
    for (int nt = 0; nt < 2; ++nt) {
      int col = n0 + wn * 32 + nt * 16 + l16;
      int rb  = m0 + wm * 64 + mt * 16 + half * 8;
      if (OUT_BF16) {
        __bf16* C = (__bf16*)Cout;
        #pragma unroll
        for (int i = 0; i < 8; ++i)
          C[(size_t)rb * DHID + (size_t)i * DHID + col] = (__bf16)acc[mt][nt][i];
      } else {
        float* C = (float*)Cout;
        #pragma unroll
        for (int i = 0; i < 8; ++i)
          C[(size_t)rb * DHID + (size_t)i * DHID + col] = acc[mt][nt][i];
      }
    }
}

// ---------------------------------------------------------------------------
// RoPE (interleaved), in-place on bf16 q/k; q scaled by 1/sqrt(dk).
// ---------------------------------------------------------------------------
__global__ void rope_kernel(__bf16* __restrict__ q, __bf16* __restrict__ k,
                            int total, float qscale) {
  int idx = blockIdx.x * blockDim.x + threadIdx.x;
  if (idx >= total) return;
  const int HALF = 64, H = 16, S = 2048;
  int i = idx % HALF;
  int h = (idx / HALF) % H;
  int s = (idx / (HALF * H)) % S;
  int row = idx / (HALF * H);                       // b*S + s
  float inv_freq = __expf(-(float)i * 0.14391156f); // 10000^(-i/64)
  float ang = (float)s * inv_freq;
  float c = cosf(ang), sn = sinf(ang);
  size_t off = (size_t)row * DHID + h * 128 + 2 * i;
  float qe = (float)q[off], qo = (float)q[off + 1];
  q[off]     = (__bf16)((qe * c - qo * sn) * qscale);
  q[off + 1] = (__bf16)((qo * c + qe * sn) * qscale);
  float ke = (float)k[off], ko = (float)k[off + 1];
  k[off]     = (__bf16)(ke * c - ko * sn);
  k[off + 1] = (__bf16)(ko * c + ke * sn);
}

// ---------------------------------------------------------------------------
// Flash attention (causal), bf16 in/out, f32 online softmax.
// S^T = K @ Q^T trick: per-lane softmax stats (one shfl_xor) and the S^T
// accumulators ARE the A-fragment of P for P@V. K/V tiles double-buffered
// through async-to-LDS; buffer set 1 reuses the (dead) Q^T staging region.
// ---------------------------------------------------------------------------
static __device__ __forceinline__ void stage_kv(const __bf16* __restrict__ Kb,
                                                const __bf16* __restrict__ Vb,
                                                int k0, __bf16* dK, __bf16* dV, int tid) {
  #pragma unroll
  for (int it = 0; it < 2; ++it) {                 // 512 x 16B chunks each
    int idx = tid + it * 256;
    int r = idx >> 4, c = idx & 15;
    copy16(Kb + (size_t)(k0 + r) * DHID + c * 8, &dK[r * 136 + c * 8]);
    copy16(Vb + (size_t)(k0 + r) * DHID + c * 8, &dV[r * 136 + c * 8]);
  }
}

// LDS element offset of K buffer for buffer set s (V buffer is +4352).
static __device__ __forceinline__ int buf_base(int s) { return s ? 0 : 17408; }

__global__ __launch_bounds__(256) void flash_attn(const __bf16* __restrict__ Q,
                                                  const __bf16* __restrict__ K,
                                                  const __bf16* __restrict__ V,
                                                  __bf16* __restrict__ O, int S) {
  // pool: [0,17408) = Q^T staging (17408 elems), later reused as buffer set 1.
  // [17408, 26112) = buffer set 0 (K tile 4352 + V tile 4352 elems).
  __shared__ __align__(16) __bf16 pool[128 * 136 + 2 * 32 * 136];  // 52224 B
  __shared__ float sBr[8][16];                                     //   512 B

  const int tid = threadIdx.x, lane = tid & 31, wave = tid >> 5;
  const int half = lane >> 4, l16 = lane & 15;
  const int q0 = blockIdx.x * 128;
  const int h = blockIdx.y, b = blockIdx.z;
  const size_t rs = DHID;
  const __bf16* Qb = Q + (size_t)b * S * rs + h * 128;
  const __bf16* Kb = K + (size_t)b * S * rs + h * 128;
  const __bf16* Vb = V + (size_t)b * S * rs + h * 128;
  __bf16*       Ob = O + (size_t)b * S * rs + h * 128;

  __bf16* sQT = pool;

  // stage Q^T once: sQT[e][m] = Q[q0+m][e]
  #pragma unroll
  for (int it = 0; it < 16; ++it) {
    int idx = tid + it * 256;             // 4096 chunks of 4 bf16
    int r = idx >> 5;                     // q row 0..127
    int c4 = (idx & 31) * 4;              // dk col
    v4bf v = *(const v4bf*)(Qb + (size_t)(q0 + r) * rs + c4);
    sQT[(c4 + 0) * 136 + r] = v[0];
    sQT[(c4 + 1) * 136 + r] = v[1];
    sQT[(c4 + 2) * 136 + r] = v[2];
    sQT[(c4 + 3) * 136 + r] = v[3];
  }
  __syncthreads();

  // B-fragments of Q^T for this wave's 16 q rows (4 dk chunks of 32)
  v16bf qfB[4];
  #pragma unroll
  for (int kc = 0; kc < 4; ++kc) {
    const __bf16* bb = &sQT[(size_t)(kc * 32 + half * 8) * 136 + wave * 16 + l16];
    #pragma unroll
    for (int s = 0; s < 16; ++s) qfB[kc][s] = bb[((s & 7) + (s >> 3) * 16) * 136];
  }
  __syncthreads();                        // Q^T region now dead -> reusable

  v8f o[8] = {};
  float m_s = -1e30f, l_s = 0.f;
  const int qrow0 = q0 + wave * 16;
  const int nIter = (q0 + 128) / 32;

  stage_kv(Kb, Vb, 0, pool + buf_base(0), pool + buf_base(0) + 4352, tid);  // tile 0

  for (int t = 0; t < nIter; ++t) {
    const int k0 = t * 32, cur = t & 1;
    if (t + 1 < nIter) {                        // prefetch next K/V tile
      int nb = buf_base(cur ^ 1);
      stage_kv(Kb, Vb, k0 + 32, pool + nb, pool + nb + 4352, tid);
      async_wait_4();                           // tile t complete (in-order)
    } else {
      async_wait_all();
    }
    __syncthreads();

    if (k0 <= qrow0 + 15) {               // wave not fully masked
      const __bf16* sK = pool + buf_base(cur);
      const __bf16* sV = sK + 4352;
      // S^T: VGPR i -> j = jt*16 + i + 8*half, lane -> m = l16
      v8f st[2] = {};
      #pragma unroll
      for (int jt = 0; jt < 2; ++jt)
        #pragma unroll
        for (int kc = 0; kc < 4; ++kc) {
          const __bf16* ab = &sK[(jt * 16 + l16) * 136 + kc * 32 + half * 8];
          v16bf kf = cat16(*(const v8bf*)ab, *(const v8bf*)(ab + 16));
          st[jt] = wmma_bf16(kf, qfB[kc], st[jt]);
        }
      // causal mask on diagonal tiles
      if (k0 + 31 > qrow0) {
        int m = qrow0 + l16;
        #pragma unroll
        for (int jt = 0; jt < 2; ++jt)
          #pragma unroll
          for (int i = 0; i < 8; ++i)
            if (k0 + jt * 16 + i + 8 * half > m) st[jt][i] = -1e30f;
      }
      // online softmax: one q row per lane, reduce across wave halves only
      float rm = -1e30f;
      #pragma unroll
      for (int i = 0; i < 8; ++i) rm = fmaxf(rm, fmaxf(st[0][i], st[1][i]));
      rm = fmaxf(rm, __shfl_xor(rm, 16, 32));
      float mn = fmaxf(m_s, rm);
      float alpha = __expf(m_s - mn);
      m_s = mn;
      float ps = 0.f;
      #pragma unroll
      for (int jt = 0; jt < 2; ++jt)
        #pragma unroll
        for (int i = 0; i < 8; ++i) { st[jt][i] = __expf(st[jt][i] - mn); ps += st[jt][i]; }
      ps += __shfl_xor(ps, 16, 32);
      l_s = l_s * alpha + ps;
      // broadcast alpha to O's row layout (row = i + 8*half) via per-wave LDS
      if (half == 0) sBr[wave][l16] = alpha;
      asm volatile("s_wait_dscnt 0" ::: "memory");
      float av[8];
      #pragma unroll
      for (int i = 0; i < 8; ++i) av[i] = sBr[wave][half * 8 + i];
      #pragma unroll
      for (int nv = 0; nv < 8; ++nv)
        #pragma unroll
        for (int i = 0; i < 8; ++i) o[nv][i] *= av[i];
      // P A-fragment is exactly the S^T accumulator pair, converted to bf16
      v16bf pf = cat16(cvt8(st[0]), cvt8(st[1]));
      #pragma unroll
      for (int nv = 0; nv < 8; ++nv) {    // O += P @ V
        const __bf16* vb = &sV[(half * 8) * 136 + nv * 16 + l16];
        v16bf vf;
        #pragma unroll
        for (int s = 0; s < 16; ++s) vf[s] = vb[((s & 7) + (s >> 3) * 16) * 136];
        o[nv] = wmma_bf16(pf, vf, o[nv]);
      }
    }
    __syncthreads();                      // buffer reusable next iteration
  }

  // normalize (row broadcast of 1/l) and store bf16
  if (half == 0) sBr[wave][l16] = 1.f / l_s;
  asm volatile("s_wait_dscnt 0" ::: "memory");
  float inv[8];
  #pragma unroll
  for (int i = 0; i < 8; ++i) inv[i] = sBr[wave][half * 8 + i];
  #pragma unroll
  for (int nv = 0; nv < 8; ++nv)
    #pragma unroll
    for (int i = 0; i < 8; ++i) {
      int row = qrow0 + i + 8 * half;
      Ob[(size_t)row * rs + nv * 16 + l16] = (__bf16)(o[nv][i] * inv[i]);
    }
}

// ---------------------------------------------------------------------------
extern "C" void kernel_launch(void* const* d_in, const int* in_sizes, int n_in,
                              void* d_out, int out_size, void* d_ws, size_t ws_size,
                              hipStream_t stream) {
  const float* x  = (const float*)d_in[0];
  const float* wq = (const float*)d_in[1];
  const float* wk = (const float*)d_in[2];
  const float* wv = (const float*)d_in[3];
  const float* wo = (const float*)d_in[4];
  float* out = (float*)d_out;

  const int B = 4, S = 2048, H = 16;
  const size_t mat  = (size_t)B * S * DHID;     // elements of [B*S, 2048]
  const size_t wmat = (size_t)DHID * DHID;

  __bf16* xb  = (__bf16*)d_ws;
  __bf16* wqT = xb  + mat;
  __bf16* wkT = wqT + wmat;
  __bf16* wvT = wkT + wmat;
  __bf16* woT = wvT + wmat;
  __bf16* qb  = woT + wmat;
  __bf16* kb  = qb  + mat;
  __bf16* vb  = kb  + mat;
  __bf16* aob = vb  + mat;

  // pre-passes: bf16 packing + weight transposes
  int n4 = (int)(mat / 4);
  cvt_f32_bf16<<<(n4 + 255) / 256, 256, 0, stream>>>(x, xb, n4);
  dim3 tg(DHID / 32, DHID / 32);
  transpose_bf16<<<tg, 256, 0, stream>>>(wq, wqT);
  transpose_bf16<<<tg, 256, 0, stream>>>(wk, wkT);
  transpose_bf16<<<tg, 256, 0, stream>>>(wv, wvT);
  transpose_bf16<<<tg, 256, 0, stream>>>(wo, woT);

  // projections (bf16 out)
  dim3 gg(DHID / 128, (B * S) / 128);           // (16, 64)
  gemm_TN<true><<<gg, 256, 0, stream>>>(xb, wqT, qb);
  gemm_TN<true><<<gg, 256, 0, stream>>>(xb, wkT, kb);
  gemm_TN<true><<<gg, 256, 0, stream>>>(xb, wvT, vb);

  int pairs = B * S * H * 64;
  rope_kernel<<<(pairs + 255) / 256, 256, 0, stream>>>(qb, kb, pairs,
                                                       0.08838834764831845f);

  flash_attn<<<dim3(S / 128, H, B), 256, 0, stream>>>(qb, kb, vb, aob, S);

  // output projection (f32 out)
  gemm_TN<false><<<gg, 256, 0, stream>>>(aob, woT, out);
}
